// Multi_atten_49280454754585
// MI455X (gfx1250) — matrix operand order repaired and checked by reading, not compiled, not run on hardware
//
#include <hip/hip_runtime.h>
#include <hip/hip_bf16.h>

// ---------------------------------------------------------------------------
// Problem constants
// ---------------------------------------------------------------------------
#define N_EMBD 1024
#define N_HEADS 16
#define DH 64
#define TSEQ 2048
#define BATCH 4
#define BT (BATCH * TSEQ)          // 8192 tokens
#define C3 (3 * N_EMBD)            // 3072

typedef __attribute__((ext_vector_type(16))) __bf16 v16bf;
typedef __attribute__((ext_vector_type(8)))  float  v8f;

// 16x32 bf16 fragment (8 VGPRs / lane)
struct Frag {
    union { v16bf v; uint4 u[2]; __bf16 h[16]; };
};
struct Pack8 {
    union { __bf16 h[8]; uint4 u; };
};

// float -> bf16, round-to-nearest-even, pure bit manipulation (storage only)
__device__ inline __bf16 f2bf(float f) {
    unsigned u = __float_as_uint(f);
    u = (u + 0x7FFFu + ((u >> 16) & 1u)) >> 16;
    unsigned short s = (unsigned short)u;
    union { unsigned short s; __bf16 b; } cvt;
    cvt.s = s;
    return cvt.b;
}

// A-matrix fragment load (16x32, row-major source, row stride in elements):
//   lane<16 : K = {0..7, 16..23},  lane>=16 : K = {8..15, 24..31}
// caller pre-offsets base by  row*ld + (hw ? 8 : 0)
__device__ inline void load_a(Frag& f, const __bf16* base) {
    f.u[0] = *(const uint4*)(base);
    f.u[1] = *(const uint4*)(base + 16);
}
// B-matrix fragment load from B^T stored row-major [N][K]:
//   lane<16 : K = {0..7, 16..23},  lane>=16 : K = {8..15, 24..31}
// caller pre-offsets base by  n*ld + (hw ? 8 : 0)
__device__ inline void load_b(Frag& f, const __bf16* base) {
    f.u[0] = *(const uint4*)(base);
    f.u[1] = *(const uint4*)(base + 16);
}

__device__ inline v8f wmma_bf16(const Frag& a, const Frag& b, v8f c) {
    return __builtin_amdgcn_wmma_f32_16x16x32_bf16(
        /*neg_a=*/false, a.v, /*neg_b=*/false, b.v,
        /*c_mod=*/(short)0, c, /*reuse_a=*/false, /*reuse_b=*/false);
}

// ---------------------------------------------------------------------------
// Elementwise conversion kernels
// ---------------------------------------------------------------------------
__global__ void k_cvt(const float* __restrict__ src, __bf16* __restrict__ dst, int n) {
    int i = blockIdx.x * blockDim.x + threadIdx.x;
    if (i < n) dst[i] = f2bf(src[i]);
}

// src[rows][cols] (row-major) -> dst[cols][rows]  (i.e. dst = src^T), bf16
__global__ void k_cvt_t(const float* __restrict__ src, __bf16* __restrict__ dst,
                        int rows, int cols) {
    int i = blockIdx.x * blockDim.x + threadIdx.x;
    if (i < rows * cols) {
        int n = i / rows;   // dst row   (source column)
        int k = i % rows;   // dst col   (source row)
        dst[i] = f2bf(src[k * cols + n]);
    }
}

// ---------------------------------------------------------------------------
// GEMM1: qkv = x @ W_attn + b_attn, scattered into Q/K (row-major [B,H,T,Dh])
// and V transposed ([B,H,Dh,T]).  One 16x16 output tile per wave.
// ---------------------------------------------------------------------------
__global__ void __launch_bounds__(256)
k_gemm_qkv(const __bf16* __restrict__ A,   // [BT][C] bf16 (x)
           const __bf16* __restrict__ Bt,  // [3C][C] bf16 (W_attn^T)
           const float*  __restrict__ bias,
           __bf16* __restrict__ Qb, __bf16* __restrict__ Kb,
           __bf16* __restrict__ Vt) {
    const int lane = threadIdx.x & 31;
    const int w    = threadIdx.x >> 5;     // 8 waves: 2 (M) x 4 (N)
    const int hw   = lane >> 4;
    const int ln   = lane & 15;
    const int rowBase = blockIdx.x * 32 + (w >> 2) * 16;
    const int colBase = blockIdx.y * 64 + (w & 3) * 16;

    const __bf16* arow = A  + (size_t)(rowBase + ln) * N_EMBD + hw * 8;
    const __bf16* brow = Bt + (size_t)(colBase + ln) * N_EMBD + hw * 8;

    v8f acc = {};
    for (int kk = 0; kk < N_EMBD; kk += 32) {
        __builtin_prefetch(arow + kk + 256, 0, 1);
        __builtin_prefetch(brow + kk + 256, 0, 1);
        Frag a, b;
        load_a(a, arow + kk);
        load_b(b, brow + kk);
        acc = wmma_bf16(a, b, acc);
    }

    // Epilogue: C/D layout -> lane holds col N=ln, rows M = r + 8*hw
    const int col = colBase + ln;
    const int mm  = col >> 10;            // 0=q, 1=k, 2=v
    const int cc  = col & (N_EMBD - 1);
    const int h   = cc >> 6;
    const int dh  = cc & (DH - 1);
    const float bv = bias[col];
    const int tBase = rowBase + hw * 8;   // 8 consecutive tokens, same batch
    const int b  = tBase >> 11;           // /TSEQ
    const int tt = tBase & (TSEQ - 1);
    const int bh = b * N_HEADS + h;

    if (mm == 2) {
        // V transposed: consecutive accumulator rows -> consecutive T: pack 8
        Pack8 p;
        #pragma unroll
        for (int r = 0; r < 8; ++r) p.h[r] = f2bf(acc[r] + bv);
        *(uint4*)(Vt + (size_t)(bh * DH + dh) * TSEQ + tt) = p.u;
    } else {
        __bf16* dst = (mm == 0 ? Qb : Kb) + (size_t)(bh * TSEQ + tt) * DH + dh;
        #pragma unroll
        for (int r = 0; r < 8; ++r) dst[(size_t)r * DH] = f2bf(acc[r] + bv);
    }
}

// ---------------------------------------------------------------------------
// Flash attention: one wave (block of 32) per 16-row Q tile of one (b,h).
// Streams 32-key tiles; online softmax; O += P @ V with V pre-transposed.
// ---------------------------------------------------------------------------
__global__ void __launch_bounds__(32)
k_attn(const __bf16* __restrict__ Qb,   // [B*H, T, Dh]
       const __bf16* __restrict__ Kb,   // [B*H, T, Dh]
       const __bf16* __restrict__ Vt,   // [B*H, Dh, T]
       __bf16* __restrict__ Obf) {      // [BT, C]
    __shared__ __align__(16) __bf16 Pl[16][32];

    const int lane  = threadIdx.x & 31;
    const int hw    = lane >> 4;
    const int ln    = lane & 15;
    const int qt    = blockIdx.x & (TSEQ / 16 - 1);
    const int bh    = blockIdx.x / (TSEQ / 16);
    const int qbase = qt * 16;

    // Q tile 16x64 -> two A fragments (Dh chunks of 32)
    Frag qa[2];
    #pragma unroll
    for (int c = 0; c < 2; ++c)
        load_a(qa[c], Qb + (size_t)(bh * TSEQ + qbase + ln) * DH + 32 * c + hw * 8);

    float mrow[8], lrow[8];
    #pragma unroll
    for (int r = 0; r < 8; ++r) { mrow[r] = -__builtin_inff(); lrow[r] = 0.0f; }
    v8f O0 = {}, O1 = {}, O2 = {}, O3 = {};

    const float scale = 0.125f;                 // 1/sqrt(64)
    const int nkt = ((qbase + 15) >> 5) + 1;    // causal key-tile count

    for (int j = 0; j < nkt; ++j) {
        const int kbase = j * 32;

        // S = Q @ K^T : two 16x16 D fragments (keys kbase..+15, +16..+31)
        v8f S0 = {}, S1 = {};
        #pragma unroll
        for (int c = 0; c < 2; ++c) {
            Frag kb0, kb1;
            load_b(kb0, Kb + (size_t)(bh * TSEQ + kbase + ln) * DH + 32 * c + hw * 8);
            load_b(kb1, Kb + (size_t)(bh * TSEQ + kbase + 16 + ln) * DH + 32 * c + hw * 8);
            S0 = wmma_bf16(qa[c], kb0, S0);
            S1 = wmma_bf16(qa[c], kb1, S1);
        }

        // scale + causal mask + online softmax (rows spread over 16-lane halves)
        #pragma unroll
        for (int r = 0; r < 8; ++r) {
            const int q = qbase + r + 8 * hw;
            float s0 = S0[r] * scale;
            float s1 = S1[r] * scale;
            if (kbase + ln > q)      s0 = -__builtin_inff();
            if (kbase + 16 + ln > q) s1 = -__builtin_inff();

            float rm = fmaxf(s0, s1);
            #pragma unroll
            for (int msk = 1; msk < 16; msk <<= 1)
                rm = fmaxf(rm, __shfl_xor(rm, msk, 32));
            const float mnew  = fmaxf(mrow[r], rm);
            const float alpha = __expf(mrow[r] - mnew);
            const float p0 = __expf(s0 - mnew);
            const float p1 = __expf(s1 - mnew);
            float rs = p0 + p1;
            #pragma unroll
            for (int msk = 1; msk < 16; msk <<= 1)
                rs += __shfl_xor(rs, msk, 32);
            lrow[r] = lrow[r] * alpha + rs;
            mrow[r] = mnew;
            O0[r] *= alpha; O1[r] *= alpha; O2[r] *= alpha; O3[r] *= alpha;

            // stage P (D layout) into LDS as bf16
            Pl[r + 8 * hw][ln]      = f2bf(p0);
            Pl[r + 8 * hw][16 + ln] = f2bf(p1);
        }

        // per-wave DS ops are in-order; wait so VALU sees staged data
        asm volatile("s_wait_dscnt 0" ::: "memory");

        // reload P in A-matrix layout (16x32)
        Frag pf;
        {
            const __bf16* pb = &Pl[ln][hw * 8];
            pf.u[0] = *(const uint4*)(pb);
            pf.u[1] = *(const uint4*)(pb + 16);
        }

        // O += P @ V : four 16x16 D fragments along Dh, B-frags from V^T rows
        Frag vf;
        load_b(vf, Vt + (size_t)(bh * DH +  0 + ln) * TSEQ + kbase + hw * 8);
        O0 = wmma_bf16(pf, vf, O0);
        load_b(vf, Vt + (size_t)(bh * DH + 16 + ln) * TSEQ + kbase + hw * 8);
        O1 = wmma_bf16(pf, vf, O1);
        load_b(vf, Vt + (size_t)(bh * DH + 32 + ln) * TSEQ + kbase + hw * 8);
        O2 = wmma_bf16(pf, vf, O2);
        load_b(vf, Vt + (size_t)(bh * DH + 48 + ln) * TSEQ + kbase + hw * 8);
        O3 = wmma_bf16(pf, vf, O3);
    }

    // normalize + write to [BT][C] bf16 for the projection GEMM
    const int b = bh / N_HEADS, h = bh % N_HEADS;
    #pragma unroll
    for (int r = 0; r < 8; ++r) {
        const float inv = 1.0f / lrow[r];
        const int tg = b * TSEQ + qbase + r + 8 * hw;
        __bf16* dst = Obf + (size_t)tg * N_EMBD + h * DH + ln;
        dst[0]  = f2bf(O0[r] * inv);
        dst[16] = f2bf(O1[r] * inv);
        dst[32] = f2bf(O2[r] * inv);
        dst[48] = f2bf(O3[r] * inv);
    }
}

// ---------------------------------------------------------------------------
// GEMM2: out = O @ W_proj + b_proj   (fp32 output)
// ---------------------------------------------------------------------------
__global__ void __launch_bounds__(256)
k_gemm_proj(const __bf16* __restrict__ A,   // [BT][C] bf16
            const __bf16* __restrict__ Bt,  // [C][C]  bf16 (W_proj^T)
            const float*  __restrict__ bias,
            float* __restrict__ out) {
    const int lane = threadIdx.x & 31;
    const int w    = threadIdx.x >> 5;
    const int hw   = lane >> 4;
    const int ln   = lane & 15;
    const int rowBase = blockIdx.x * 32 + (w >> 2) * 16;
    const int colBase = blockIdx.y * 64 + (w & 3) * 16;

    const __bf16* arow = A  + (size_t)(rowBase + ln) * N_EMBD + hw * 8;
    const __bf16* brow = Bt + (size_t)(colBase + ln) * N_EMBD + hw * 8;

    v8f acc = {};
    for (int kk = 0; kk < N_EMBD; kk += 32) {
        __builtin_prefetch(arow + kk + 256, 0, 1);
        __builtin_prefetch(brow + kk + 256, 0, 1);
        Frag a, b;
        load_a(a, arow + kk);
        load_b(b, brow + kk);
        acc = wmma_bf16(a, b, acc);
    }

    const int col = colBase + ln;
    const float bv = bias[col];
    const int tBase = rowBase + hw * 8;
    #pragma unroll
    for (int r = 0; r < 8; ++r)
        out[(size_t)(tBase + r) * N_EMBD + col] = acc[r] + bv;
}

// ---------------------------------------------------------------------------
// Launcher
// ---------------------------------------------------------------------------
extern "C" void kernel_launch(void* const* d_in, const int* in_sizes, int n_in,
                              void* d_out, int out_size, void* d_ws, size_t ws_size,
                              hipStream_t stream) {
    (void)in_sizes; (void)n_in; (void)out_size; (void)ws_size;
    const float* x      = (const float*)d_in[0];
    const float* W_attn = (const float*)d_in[1];
    const float* b_attn = (const float*)d_in[2];
    const float* W_proj = (const float*)d_in[3];
    const float* b_proj = (const float*)d_in[4];
    float* out = (float*)d_out;

    char* ws = (char*)d_ws;
    // byte offsets (all 256B-aligned); total ~92.3 MB
    __bf16* xbf = (__bf16*)(ws + 0);                         // 16 MB  [BT][C]
    __bf16* WaT = (__bf16*)(ws + 16777216);                  //  6 MB  [3C][C]
    __bf16* WpT = (__bf16*)(ws + 16777216 + 6291456);        //  2 MB  [C][C]
    __bf16* Qb  = (__bf16*)(ws + 25165824);                  // 16 MB  [B*H,T,Dh]
    __bf16* Kb  = (__bf16*)(ws + 41943040);                  // 16 MB  [B*H,T,Dh]
    __bf16* Vt  = (__bf16*)(ws + 58720256);                  // 16 MB  [B*H,Dh,T]
    __bf16* Obf = (__bf16*)(ws + 75497472);                  // 16 MB  [BT][C]

    k_cvt  <<<(BT * N_EMBD) / 256, 256, 0, stream>>>(x, xbf, BT * N_EMBD);
    k_cvt_t<<<(N_EMBD * C3) / 256, 256, 0, stream>>>(W_attn, WaT, N_EMBD, C3);
    k_cvt_t<<<(N_EMBD * N_EMBD) / 256, 256, 0, stream>>>(W_proj, WpT, N_EMBD, N_EMBD);

    k_gemm_qkv<<<dim3(BT / 32, C3 / 64), 256, 0, stream>>>(xbf, WaT, b_attn, Qb, Kb, Vt);

    k_attn<<<BATCH * N_HEADS * (TSEQ / 16), 32, 0, stream>>>(Qb, Kb, Vt, Obf);

    k_gemm_proj<<<dim3(BT / 32, N_EMBD / 64), 256, 0, stream>>>(Obf, WpT, b_proj, out);
}
